// HierarchosCore_38843684225193
// MI455X (gfx1250) — matrix operand order repaired and checked by reading, not compiled, not run on hardware
//
#include <hip/hip_runtime.h>

typedef __attribute__((ext_vector_type(16))) __bf16 v16bf;
typedef __attribute__((ext_vector_type(8)))  float  v8f;
typedef __attribute__((ext_vector_type(4)))  unsigned int v4u;
typedef unsigned short u16;

#define NB 512
#define ND 1024
#define NH 2048
#define KDIM 2048          // inner dim of every GEMM (2*D == H == 2048)
#define NSTEPS 8
#define C_ATOL 0.01f
#define C_THR  0.1f

#define BH (NB*NH)         // 1048576
#define BD (NB*ND)         // 524288
#define OUT_ENC    0
#define OUT_STATE  (NB*ND)                       // 524288
#define OUT_COMMIT (OUT_STATE + (size_t)NB*NH*5) // 5767168
#define OUT_DRIFT  (OUT_COMMIT + 1)              // 5767169

__device__ __forceinline__ u16 f2bf(float f) {
  unsigned u = __float_as_uint(f);
  u += 0x7fffu + ((u >> 16) & 1u);               // round-to-nearest-even
  return (u16)(u >> 16);
}
__device__ __forceinline__ float clip50(float v){ return fminf(fmaxf(v, -50.f), 50.f); }
__device__ __forceinline__ float sigm(float x){ return 1.f / (1.f + expf(-x)); }

union FragAB { v4u q[2]; v16bf v; };

// ---------------- WMMA bf16 GEMM:  C[M,N](f32) = A[M,K](bf16) * WT[N,K](bf16)^T ----
// MODE 0: C = acc
// MODE 1: C = clip(acc + bias[n], +-50)          (l_input_proj)
// MODE 2: C = acc + bias[n] + res[m,n]           (final_enc = enc + out@Wout + b_out)
template<int MODE>
__global__ __launch_bounds__(128)
void gemm_bf16(const u16* __restrict__ A, const u16* __restrict__ WT,
               float* __restrict__ C, const float* __restrict__ bias,
               const float* __restrict__ res, int M, int N)
{
  const int lane = threadIdx.x & 31;
  const int wave = threadIdx.x >> 5;
  const int lo   = lane & 15;
  const int hi   = lane >> 4;
  const int n0   = blockIdx.x * 128 + wave * 32;
  const int m0   = blockIdx.y * 64;

  v8f acc[4][2];
  v8f zero = {0.f,0.f,0.f,0.f,0.f,0.f,0.f,0.f};
#pragma unroll
  for (int mt = 0; mt < 4; ++mt)
#pragma unroll
    for (int nt = 0; nt < 2; ++nt) acc[mt][nt] = zero;

  const u16* aRow[4];
#pragma unroll
  for (int mt = 0; mt < 4; ++mt)
    aRow[mt] = A + (size_t)(m0 + 16*mt + lo) * KDIM + 16*hi;
  const u16* bRow[2];
#pragma unroll
  for (int nt = 0; nt < 2; ++nt)
    bRow[nt] = WT + (size_t)(n0 + 16*nt + lo) * KDIM + 16*hi;

  for (int kk = 0; kk < KDIM; kk += 32) {
    FragAB a[4], b[2];
#pragma unroll
    for (int mt = 0; mt < 4; ++mt) {            // A 16x32: K[k+16h..+16) contiguous
      const u16* p = aRow[mt] + kk;
      a[mt].q[0] = *(const v4u*)(p);
      a[mt].q[1] = *(const v4u*)(p + 8);
    }
#pragma unroll
    for (int nt = 0; nt < 2; ++nt) {            // B 32x16: 32 contiguous bytes of W^T row
      const u16* p = bRow[nt] + kk;
      b[nt].q[0] = *(const v4u*)(p);
      b[nt].q[1] = *(const v4u*)(p + 8);
    }
#pragma unroll
    for (int mt = 0; mt < 4; ++mt)
#pragma unroll
      for (int nt = 0; nt < 2; ++nt)
        acc[mt][nt] = __builtin_amdgcn_wmma_f32_16x16x32_bf16(
            false, a[mt].v, false, b[nt].v, (short)0, acc[mt][nt], false, false);
  }

#pragma unroll
  for (int mt = 0; mt < 4; ++mt)
#pragma unroll
    for (int nt = 0; nt < 2; ++nt) {
      int col = n0 + 16*nt + lo;
#pragma unroll
      for (int r = 0; r < 8; ++r) {             // C: VGPR r -> M = r + 8*hi, N = lo
        int row = m0 + 16*mt + r + 8*hi;
        float v = acc[mt][nt][r];
        if (MODE == 1) v = clip50(v + bias[col]);
        if (MODE == 2) v = v + bias[col] + res[(size_t)row * N + col];
        C[(size_t)row * N + col] = v;
      }
    }
}

// ---------------- weight convert+transpose: W[K,N] f32 -> WT[N,K] bf16 -------------
__global__ __launch_bounds__(256)
void k_wcvt(const float* __restrict__ W, u16* __restrict__ WT, int N, long total)
{
  long i = (long)blockIdx.x * 256 + threadIdx.x;
  if (i >= total) return;
  int n = (int)(i / KDIM);
  int k = (int)(i % KDIM);
  WT[i] = f2bf(W[(size_t)k * N + n]);
}

// ---------------- init: planar shadow state + drift + scalars ----------------------
__global__ __launch_bounds__(256)
void k_init(const float* __restrict__ lst, const float* __restrict__ idrift,
            float* sxx, float* saa, float* sbb, float* spp, float* scc,
            float* drift, float* scal, int* flags)
{
  int i = blockIdx.x * 256 + threadIdx.x;
  if (i < BH) {
    size_t b = (size_t)i * 5;
    sxx[i] = lst[b+0]; saa[i] = lst[b+1]; sbb[i] = lst[b+2];
    spp[i] = lst[b+3]; scc[i] = lst[b+4];
  }
  if (i < BD) drift[i] = idrift[i];
  if (i == 0) { scal[0]=0.f; scal[1]=0.f; scal[2]=0.f; scal[3]=0.f; flags[0]=0; flags[1]=1; }
}

// ---------------- A = bf16(concat[enc, sc + drift]) --------------------------------
__global__ __launch_bounds__(256)
void k_buildA(const float* __restrict__ enc, const float* __restrict__ sc,
              const float* __restrict__ drift, u16* __restrict__ Abf)
{
  int i = blockIdx.x * 256 + threadIdx.x;
  if (i >= NB * 2 * ND) return;
  int b = i >> 11;
  int j = i & 2047;
  float v = (j < ND) ? enc[(size_t)b * ND + j]
                     : sc[(size_t)b * ND + (j - ND)] + drift[(size_t)b * ND + (j - ND)];
  Abf[i] = f2bf(v);
}

// ---------------- all five time/channel mixes --------------------------------------
__global__ __launch_bounds__(256)
void k_mix(const float* __restrict__ li,
           const float* __restrict__ xxp, int xs,
           const float* __restrict__ ccp, int cs,
           const float* __restrict__ mk,  const float* __restrict__ mv,
           const float* __restrict__ mr,  const float* __restrict__ mk2,
           const float* __restrict__ mr2,
           u16* xk, u16* xv, u16* xr, u16* xk2, u16* xr2)
{
  int i = blockIdx.x * 256 + threadIdx.x;
  if (i >= BH) return;
  int h = i & (NH - 1);
  float x  = li[i];
  float xx = xxp[(size_t)i * xs];
  float cc = ccp[(size_t)i * cs];
  xk [i] = f2bf(x * mk [h] + xx * (1.f - mk [h]));
  xv [i] = f2bf(x * mv [h] + xx * (1.f - mv [h]));
  xr [i] = f2bf(x * mr [h] + xx * (1.f - mr [h]));
  xk2[i] = f2bf(x * mk2[h] + cc * (1.f - mk2[h]));
  xr2[i] = f2bf(x * mr2[h] + cc * (1.f - mr2[h]));
}

// ---------------- WKV core + candidate next state ----------------------------------
__global__ __launch_bounds__(256)
void k_wkv(const float* __restrict__ li,
           const float* __restrict__ kb, const float* __restrict__ vb,
           const float* __restrict__ rpre,
           const float* __restrict__ aap, const float* __restrict__ bbp,
           const float* __restrict__ ppp, int ss,
           const float* __restrict__ tf, const float* __restrict__ td,
           u16* __restrict__ a_bf,
           float* cxx, float* caa, float* cbb, float* cpp, float* ccc,
           float* st_out)
{
  int i = blockIdx.x * 256 + threadIdx.x;
  if (i >= BH) return;
  int h = i & (NH - 1);
  float k  = kb[i], v = vb[i];
  float r  = sigm(rpre[i]);
  float p  = ppp[(size_t)i * ss];
  float A  = aap[(size_t)i * ss];
  float Bb = bbp[(size_t)i * ss];
  float x  = li[i];
  float ww = tf[h] + k;
  float q  = fmaxf(p, ww);
  float e1 = expf(p - q), e2 = expf(ww - q);
  float wkv = (e1 * A + e2 * v) / (e1 * Bb + e2);
  float ww2 = p - expf(td[h]);
  float q2  = fmaxf(ww2, k);
  float e1b = expf(ww2 - q2), e2b = expf(k - q2);
  a_bf[i] = f2bf(r * wkv);
  float nxx = clip50(x);
  float naa = clip50(e1b * A + e2b * v);
  float nbb = clip50(e1b * Bb + e2b);
  float npp = clip50(q2);
  if (st_out) {                                  // final commit: interleaved [B,H,5]
    size_t b = (size_t)i * 5;
    st_out[b+0] = nxx; st_out[b+1] = naa; st_out[b+2] = nbb;
    st_out[b+3] = npp; st_out[b+4] = nxx;
  } else {
    cxx[i] = nxx; caa[i] = naa; cbb[i] = nbb; cpp[i] = npp; ccc[i] = nxx;
  }
}

__global__ __launch_bounds__(256)
void k_kk(const float* __restrict__ kkpre, u16* __restrict__ kkbf)
{
  int i = blockIdx.x * 256 + threadIdx.x;
  if (i >= BH) return;
  float t = fmaxf(kkpre[i], 0.f);
  kkbf[i] = f2bf(t * t);
}

__global__ __launch_bounds__(256)
void k_out(const float* __restrict__ tm, const float* __restrict__ r2pre,
           const float* __restrict__ cm, u16* __restrict__ out_bf)
{
  int i = blockIdx.x * 256 + threadIdx.x;
  if (i >= BH) return;
  out_bf[i] = f2bf(tm[i] + sigm(r2pre[i]) * cm[i]);
}

__global__ void k_zero2(float* scal) { scal[0] = 0.f; scal[1] = 0.f; }

// delta = tanh(dd); nd = clip(drift+delta); reduce sum(nd^2) and sum|delta|
__global__ __launch_bounds__(256)
void k_delta(const float* __restrict__ dd, const float* __restrict__ drift,
             float* __restrict__ nd, float* __restrict__ scal)
{
  __shared__ float sh0[256], sh1[256];
  int i = blockIdx.x * 256 + threadIdx.x;
  float p0 = 0.f, p1 = 0.f;
  if (i < BD) {
    float d = tanhf(dd[i]);
    float v = fminf(fmaxf(drift[i] + d, -5.f), 5.f);
    nd[i] = v; p0 = v * v; p1 = fabsf(d);
  }
  sh0[threadIdx.x] = p0; sh1[threadIdx.x] = p1;
  __syncthreads();
  for (int s = 128; s > 0; s >>= 1) {
    if (threadIdx.x < s) { sh0[threadIdx.x] += sh0[threadIdx.x + s];
                           sh1[threadIdx.x] += sh1[threadIdx.x + s]; }
    __syncthreads();
  }
  if (threadIdx.x == 0) { atomicAdd(&scal[0], sh0[0]); atomicAdd(&scal[1], sh1[0]); }
}

__global__ void k_step_update(float* scal, int* flags)
{
  float hinge = fminf(fmaxf(scal[0] / (float)NB - C_THR, 0.f), 100.f);
  int conv = (scal[1] / (float)BD) < C_ATOL;
  int d = flags[0];
  flags[1] = !d;                                 // commit this step iff not yet done
  if (!d) { scal[2] += hinge; scal[3] += 1.f; }
  if (conv) flags[0] = 1;
}

__global__ __launch_bounds__(256)
void k_commit_state(const int* __restrict__ flags,
                    const float* cxx, const float* caa, const float* cbb,
                    const float* cpp, const float* ccc,
                    float* sxx, float* saa, float* sbb, float* spp, float* scc)
{
  if (!flags[1]) return;
  int i = blockIdx.x * 256 + threadIdx.x;
  if (i >= BH) return;
  sxx[i] = cxx[i]; saa[i] = caa[i]; sbb[i] = cbb[i]; spp[i] = cpp[i]; scc[i] = ccc[i];
}

__global__ __launch_bounds__(256)
void k_commit_drift(const int* __restrict__ flags,
                    const float* __restrict__ nd, float* __restrict__ drift)
{
  if (!flags[1]) return;
  int i = blockIdx.x * 256 + threadIdx.x;
  if (i >= BD) return;
  drift[i] = nd[i];
}

__global__ void k_finalize(const float* scal, float* out)
{ out[0] = scal[2] / fmaxf(scal[3], 1.f); }

__global__ __launch_bounds__(256)
void k_copydrift(const float* __restrict__ drift, float* __restrict__ out)
{
  int i = blockIdx.x * 256 + threadIdx.x;
  if (i < BD) out[i] = drift[i];
}

// -----------------------------------------------------------------------------------
static void launch_gemm(int mode, const u16* A, const u16* WT, float* C,
                        const float* bias, const float* res, int M, int N,
                        hipStream_t s)
{
  dim3 g(N / 128, M / 64), b(128);
  if (mode == 0)      gemm_bf16<0><<<g, b, 0, s>>>(A, WT, C, bias, res, M, N);
  else if (mode == 1) gemm_bf16<1><<<g, b, 0, s>>>(A, WT, C, bias, res, M, N);
  else                gemm_bf16<2><<<g, b, 0, s>>>(A, WT, C, bias, res, M, N);
}

extern "C" void kernel_launch(void* const* d_in, const int* in_sizes, int n_in,
                              void* d_out, int out_size, void* d_ws, size_t ws_size,
                              hipStream_t stream)
{
  (void)in_sizes; (void)n_in; (void)out_size; (void)ws_size;
  const float* enc    = (const float*)d_in[0];
  const float* sc     = (const float*)d_in[1];
  const float* lst    = (const float*)d_in[2];
  const float* idrift = (const float*)d_in[3];
  const float* Win    = (const float*)d_in[4];
  const float* b_in   = (const float*)d_in[5];
  const float* Wdrift = (const float*)d_in[6];
  const float* Wout   = (const float*)d_in[7];
  const float* b_out  = (const float*)d_in[8];
  const float* mix_k  = (const float*)d_in[9];
  const float* mix_v  = (const float*)d_in[10];
  const float* mix_r  = (const float*)d_in[11];
  const float* mix_k2 = (const float*)d_in[12];
  const float* mix_r2 = (const float*)d_in[13];
  const float* tdcy   = (const float*)d_in[14];
  const float* tfst   = (const float*)d_in[15];
  const float* Wk  = (const float*)d_in[16];
  const float* Wv  = (const float*)d_in[17];
  const float* Wr  = (const float*)d_in[18];
  const float* Wo  = (const float*)d_in[19];
  const float* Wk2 = (const float*)d_in[20];
  const float* Wv2 = (const float*)d_in[21];
  const float* Wr2 = (const float*)d_in[22];
  float* fout = (float*)d_out;

  // ---- workspace bump allocator (256B aligned) ----
  char* wp = (char*)d_ws;
  auto alloc = [&](size_t bytes) -> char* {
    char* r = wp; wp += (bytes + 255) & ~(size_t)255; return r;
  };
  const size_t WSQ = sizeof(u16) * (size_t)NH * KDIM;   // 2048x2048 bf16
  const size_t WRC = sizeof(u16) * (size_t)ND * KDIM;   // 1024x2048 bf16
  u16* WinT  = (u16*)alloc(WSQ);
  u16* WkT   = (u16*)alloc(WSQ);
  u16* WvT   = (u16*)alloc(WSQ);
  u16* WrT   = (u16*)alloc(WSQ);
  u16* WoT   = (u16*)alloc(WSQ);
  u16* Wk2T  = (u16*)alloc(WSQ);
  u16* Wv2T  = (u16*)alloc(WSQ);
  u16* Wr2T  = (u16*)alloc(WSQ);
  u16* WdrT  = (u16*)alloc(WRC);
  u16* WouT  = (u16*)alloc(WRC);
  u16* Abf   = (u16*)alloc(sizeof(u16) * (size_t)BH);
  u16* xk    = (u16*)alloc(sizeof(u16) * (size_t)BH);
  u16* xv    = (u16*)alloc(sizeof(u16) * (size_t)BH);
  u16* xr    = (u16*)alloc(sizeof(u16) * (size_t)BH);
  u16* xk2   = (u16*)alloc(sizeof(u16) * (size_t)BH);
  u16* xr2   = (u16*)alloc(sizeof(u16) * (size_t)BH);
  u16* a_bf  = (u16*)alloc(sizeof(u16) * (size_t)BH);
  u16* kk_bf = (u16*)alloc(sizeof(u16) * (size_t)BH);
  u16* out_bf= (u16*)alloc(sizeof(u16) * (size_t)BH);
  float* li  = (float*)alloc(sizeof(float) * (size_t)BH);
  float* kbuf= (float*)alloc(sizeof(float) * (size_t)BH);
  float* vbuf= (float*)alloc(sizeof(float) * (size_t)BH);
  float* rpre= (float*)alloc(sizeof(float) * (size_t)BH);
  float* cxx = (float*)alloc(sizeof(float) * (size_t)BH);
  float* caa = (float*)alloc(sizeof(float) * (size_t)BH);
  float* cbb = (float*)alloc(sizeof(float) * (size_t)BH);
  float* cpp = (float*)alloc(sizeof(float) * (size_t)BH);
  float* ccc = (float*)alloc(sizeof(float) * (size_t)BH);
  float* sxx = (float*)alloc(sizeof(float) * (size_t)BH);
  float* saa = (float*)alloc(sizeof(float) * (size_t)BH);
  float* sbb = (float*)alloc(sizeof(float) * (size_t)BH);
  float* spp = (float*)alloc(sizeof(float) * (size_t)BH);
  float* scc = (float*)alloc(sizeof(float) * (size_t)BH);
  float* dd  = (float*)alloc(sizeof(float) * (size_t)BD);
  float* nd  = (float*)alloc(sizeof(float) * (size_t)BD);
  float* drift = (float*)alloc(sizeof(float) * (size_t)BD);
  float* scal  = (float*)alloc(sizeof(float) * 4);
  int*   flags = (int*)alloc(sizeof(int) * 2);
  // buffer reuse (safe: producer/consumer strictly ordered on stream)
  float* tm    = kbuf;   // Wo output after kbuf consumed by k_wkv
  float* kkpre = vbuf;   // Wk2 output after vbuf consumed
  float* r2pre = rpre;   // Wr2 output after rpre consumed
  float* cm    = li;     // Wv2 output after li consumed

  const int GBH = (BH + 255) / 256;
  const int GBD = (BD + 255) / 256;
  const long TSQ = (long)NH * KDIM;
  const long TRC = (long)ND * KDIM;
  const int GSQ = (int)((TSQ + 255) / 256);
  const int GRC = (int)((TRC + 255) / 256);

  // ---- per-launch weight convert + transpose to bf16 (stays L2-resident) ----
  k_wcvt<<<GSQ, 256, 0, stream>>>(Win, WinT, NH, TSQ);
  k_wcvt<<<GSQ, 256, 0, stream>>>(Wk,  WkT,  NH, TSQ);
  k_wcvt<<<GSQ, 256, 0, stream>>>(Wv,  WvT,  NH, TSQ);
  k_wcvt<<<GSQ, 256, 0, stream>>>(Wr,  WrT,  NH, TSQ);
  k_wcvt<<<GSQ, 256, 0, stream>>>(Wo,  WoT,  NH, TSQ);
  k_wcvt<<<GSQ, 256, 0, stream>>>(Wk2, Wk2T, NH, TSQ);
  k_wcvt<<<GSQ, 256, 0, stream>>>(Wv2, Wv2T, NH, TSQ);
  k_wcvt<<<GSQ, 256, 0, stream>>>(Wr2, Wr2T, NH, TSQ);
  k_wcvt<<<GRC, 256, 0, stream>>>(Wdrift, WdrT, ND, TRC);
  k_wcvt<<<GRC, 256, 0, stream>>>(Wout,   WouT, ND, TRC);

  k_init<<<GBH, 256, 0, stream>>>(lst, idrift, sxx, saa, sbb, spp, scc,
                                  drift, scal, flags);

  // ---- 8 scan steps (freeze-on-converged semantics via flags) ----
  for (int s = 0; s < NSTEPS; ++s) {
    k_buildA<<<GBH, 256, 0, stream>>>(enc, sc, drift, Abf);
    launch_gemm(1, Abf, WinT, li, b_in, nullptr, NB, NH, stream);
    k_mix<<<GBH, 256, 0, stream>>>(li, sxx, 1, scc, 1,
                                   mix_k, mix_v, mix_r, mix_k2, mix_r2,
                                   xk, xv, xr, xk2, xr2);
    launch_gemm(0, xk, WkT, kbuf, nullptr, nullptr, NB, NH, stream);
    launch_gemm(0, xv, WvT, vbuf, nullptr, nullptr, NB, NH, stream);
    launch_gemm(0, xr, WrT, rpre, nullptr, nullptr, NB, NH, stream);
    k_wkv<<<GBH, 256, 0, stream>>>(li, kbuf, vbuf, rpre, saa, sbb, spp, 1,
                                   tfst, tdcy, a_bf, cxx, caa, cbb, cpp, ccc,
                                   nullptr);
    launch_gemm(0, a_bf, WoT,  tm,    nullptr, nullptr, NB, NH, stream);
    launch_gemm(0, xk2,  Wk2T, kkpre, nullptr, nullptr, NB, NH, stream);
    launch_gemm(0, xr2,  Wr2T, r2pre, nullptr, nullptr, NB, NH, stream);
    k_kk<<<GBH, 256, 0, stream>>>(kkpre, kk_bf);
    launch_gemm(0, kk_bf, Wv2T, cm, nullptr, nullptr, NB, NH, stream);
    k_out<<<GBH, 256, 0, stream>>>(tm, r2pre, cm, out_bf);
    launch_gemm(0, out_bf, WdrT, dd, nullptr, nullptr, NB, ND, stream);
    k_zero2<<<1, 1, 0, stream>>>(scal);
    k_delta<<<GBD, 256, 0, stream>>>(dd, drift, nd, scal);
    k_step_update<<<1, 1, 0, stream>>>(scal, flags);
    k_commit_state<<<GBH, 256, 0, stream>>>(flags, cxx, caa, cbb, cpp, ccc,
                                            sxx, saa, sbb, spp, scc);
    k_commit_drift<<<GBD, 256, 0, stream>>>(flags, nd, drift);
  }

  // ---- final commit step with ORIGINAL l_state (stride-5 reads) ----
  k_buildA<<<GBH, 256, 0, stream>>>(enc, sc, drift, Abf);
  launch_gemm(1, Abf, WinT, li, b_in, nullptr, NB, NH, stream);
  k_mix<<<GBH, 256, 0, stream>>>(li, lst + 0, 5, lst + 4, 5,
                                 mix_k, mix_v, mix_r, mix_k2, mix_r2,
                                 xk, xv, xr, xk2, xr2);
  launch_gemm(0, xk, WkT, kbuf, nullptr, nullptr, NB, NH, stream);
  launch_gemm(0, xv, WvT, vbuf, nullptr, nullptr, NB, NH, stream);
  launch_gemm(0, xr, WrT, rpre, nullptr, nullptr, NB, NH, stream);
  k_wkv<<<GBH, 256, 0, stream>>>(li, kbuf, vbuf, rpre, lst + 1, lst + 2, lst + 3, 5,
                                 tfst, tdcy, a_bf,
                                 nullptr, nullptr, nullptr, nullptr, nullptr,
                                 fout + OUT_STATE);        // next_state -> d_out
  launch_gemm(0, a_bf, WoT,  tm,    nullptr, nullptr, NB, NH, stream);
  launch_gemm(0, xk2,  Wk2T, kkpre, nullptr, nullptr, NB, NH, stream);
  launch_gemm(0, xr2,  Wr2T, r2pre, nullptr, nullptr, NB, NH, stream);
  k_kk<<<GBH, 256, 0, stream>>>(kkpre, kk_bf);
  launch_gemm(0, kk_bf, Wv2T, cm, nullptr, nullptr, NB, NH, stream);
  k_out<<<GBH, 256, 0, stream>>>(tm, r2pre, cm, out_bf);
  // final_enc = enc + out @ Wout + b_out  -> d_out[0 .. B*D)
  launch_gemm(2, out_bf, WouT, fout + OUT_ENC, b_out, enc, NB, ND, stream);
  k_finalize<<<1, 1, 0, stream>>>(scal, fout + OUT_COMMIT);
  k_copydrift<<<GBD, 256, 0, stream>>>(drift, fout + OUT_DRIFT);
}
